// BulkSpaceGenerator_6193342841125
// MI455X (gfx1250) — compile-verified, matrix operand order checked by reading, not run on hardware
//
#include <hip/hip_runtime.h>

#define D_MODEL   1024
#define BULK_DIM  10
#define M_TOTAL   8192      // B*N = 4*2048

typedef __attribute__((ext_vector_type(2)))  float          v2f;
typedef __attribute__((ext_vector_type(4)))  float          v4f;
typedef __attribute__((ext_vector_type(8)))  float          v8f;
typedef __attribute__((ext_vector_type(8)))  __bf16         v8bf;
typedef __attribute__((ext_vector_type(16))) __bf16         v16bf;
typedef __attribute__((ext_vector_type(8)))  unsigned short v8u16;

// ---- scan-collapse weights: w[j] = (1/10) * sum_{k=j..9} 1/(k+1) ----------
__device__ __forceinline__ void fold_weights(float w[BULK_DIM]) {
    float h = 0.0f;
#pragma unroll
    for (int j = BULK_DIM - 1; j >= 0; --j) {
        h += 1.0f / (float)(j + 1);
        w[j] = 0.1f * h;
    }
}

// ---- bf16 round-to-nearest-even split helpers -----------------------------
__device__ __forceinline__ unsigned short f32_to_bf16(float x) {
    unsigned int u = __builtin_bit_cast(unsigned int, x);
    unsigned int r = u + 0x7fffu + ((u >> 16) & 1u);
    return (unsigned short)(r >> 16);
}
__device__ __forceinline__ float bf16_to_f32(unsigned short h) {
    unsigned int u = ((unsigned int)h) << 16;
    return __builtin_bit_cast(float, u);
}

// ===========================================================================
// Kernel 1: fold W (1024 x 10240) -> W_eff, split into bf16 hi/lo, and store
// TRANSPOSED (n-major: WT[d][m]) so the GEMM stages B with contiguous b128s.
// 64x64 tile per block, transpose via LDS.
// ===========================================================================
__global__ __launch_bounds__(256) void fold_split_w_kernel(
    const float* __restrict__ W, unsigned short* __restrict__ WhiT,
    unsigned short* __restrict__ WloT) {
    __shared__ float T[64 * 65];
    float w[BULK_DIM];
    fold_weights(w);
    const int tid = threadIdx.x;
    const int d0 = blockIdx.x * 64;
    const int m0 = blockIdx.y * 64;
    // phase 1: compute W_eff tile (coalesced along d) into LDS
#pragma unroll
    for (int p = 0; p < 16; ++p) {
        int e = tid + p * 256;
        int r = e >> 6, c = e & 63;
        float acc = 0.0f;
#pragma unroll
        for (int j = 0; j < BULK_DIM; ++j)
            acc += w[j] * W[(size_t)(m0 + r) * (D_MODEL * BULK_DIM) + j * D_MODEL + d0 + c];
        T[r * 65 + c] = acc;
    }
    __syncthreads();
    // phase 2: transposed, split, b128 stores (coalesced along m)
#pragma unroll
    for (int p = 0; p < 2; ++p) {
        int s = tid + p * 256;
        int r2 = s >> 3, c8 = s & 7;
        v8u16 hi, lo;
#pragma unroll
        for (int i = 0; i < 8; ++i) {
            float v = T[(c8 * 8 + i) * 65 + r2];
            unsigned short h = f32_to_bf16(v);
            hi[i] = h;
            lo[i] = f32_to_bf16(v - bf16_to_f32(h));
        }
        size_t dst = (size_t)(d0 + r2) * D_MODEL + m0 + c8 * 8;
        *(v8u16*)(WhiT + dst) = hi;
        *(v8u16*)(WloT + dst) = lo;
    }
}

// ===========================================================================
// Kernel 2: split X (8192x1024 fp32) into bf16 hi/lo planes. Pure bandwidth.
// ===========================================================================
__global__ __launch_bounds__(256) void split_x_kernel(
    const float* __restrict__ X, unsigned short* __restrict__ Xhi,
    unsigned short* __restrict__ Xlo) {
    size_t base = ((size_t)blockIdx.x * 256 + threadIdx.x) * 8;
    v4f a = *(const v4f*)(X + base);
    v4f b = *(const v4f*)(X + base + 4);
    v8u16 hi, lo;
#pragma unroll
    for (int i = 0; i < 8; ++i) {
        float x = (i < 4) ? a[i] : b[i - 4];
        unsigned short h = f32_to_bf16(x);
        hi[i] = h;
        lo[i] = f32_to_bf16(x - bf16_to_f32(h));
    }
    *(v8u16*)(Xhi + base) = hi;
    *(v8u16*)(Xlo + base) = lo;
}

// ===========================================================================
// Kernel 3: Out = X @ W_eff + b_eff via bf16x3 split WMMA (f32 accumulate).
// 8 waves/WG, 128x64 WG tile, per-wave 2x2 of 16x16 tiles, K chunks of 64.
// ===========================================================================
#define KC2 64
#define LST 72    // LDS row stride in bf16 units (16B-aligned, conflict-free)

__device__ __forceinline__ v16bf ld_frag(const unsigned short* S, int row, int koff) {
    // ISA 16-bit A/B layout: lane-half h holds K = {h*8..h*8+7, 16+h*8..23+h*8}
    v8bf a = *(const v8bf*)(S + row * LST + koff);
    v8bf b = *(const v8bf*)(S + row * LST + koff + 16);
    return __builtin_shufflevector(a, b, 0, 1, 2, 3, 4, 5, 6, 7,
                                   8, 9, 10, 11, 12, 13, 14, 15);
}

__global__ __launch_bounds__(256) void gemm_bf16x3_kernel(
    const unsigned short* __restrict__ Xhi, const unsigned short* __restrict__ Xlo,
    const unsigned short* __restrict__ WhiT, const unsigned short* __restrict__ WloT,
    const float* __restrict__ bvec, float* __restrict__ Out) {

    __shared__ unsigned short XhiS[128 * LST], XloS[128 * LST];
    __shared__ unsigned short WhiS[64 * LST],  WloS[64 * LST];

    const int tid  = threadIdx.x;
    const int lane = tid & 31;
    const int wave = tid >> 5;
    const int wm   = wave & 3;          // wave grid 4 (M) x 2 (N)
    const int wn   = wave >> 2;
    const int m0   = blockIdx.x * 128;
    const int n0   = blockIdx.y * 64;
    const int l15  = lane & 15;
    const int h8   = (lane >> 4) << 3;  // lane-half K offset

    v8f c00 = {}, c01 = {}, c10 = {}, c11 = {};

    for (int k0 = 0; k0 < D_MODEL; k0 += KC2) {
        // ---- stage X hi/lo tiles: 128x64 bf16 each, b128 copies ----
#pragma unroll
        for (int i = 0; i < 4; ++i) {
            int idx = tid + i * 256;
            int row = idx >> 3, c8 = idx & 7;
            size_t src = (size_t)(m0 + row) * D_MODEL + k0 + c8 * 8;
            *(v8u16*)(&XhiS[row * LST + c8 * 8]) = *(const v8u16*)(Xhi + src);
            *(v8u16*)(&XloS[row * LST + c8 * 8]) = *(const v8u16*)(Xlo + src);
        }
        // ---- stage W hi/lo tiles (already transposed): 64x64 bf16 each ----
#pragma unroll
        for (int i = 0; i < 2; ++i) {
            int idx = tid + i * 256;
            int row = idx >> 3, c8 = idx & 7;
            size_t src = (size_t)(n0 + row) * D_MODEL + k0 + c8 * 8;
            *(v8u16*)(&WhiS[row * LST + c8 * 8]) = *(const v8u16*)(WhiT + src);
            *(v8u16*)(&WloS[row * LST + c8 * 8]) = *(const v8u16*)(WloT + src);
        }
        if (k0 + KC2 < D_MODEL) {  // global_prefetch_b8 for next chunk
            size_t pofs = (size_t)(m0 + (tid >> 1)) * D_MODEL + k0 + KC2 + (tid & 1) * 32;
            __builtin_prefetch(Xhi + pofs, 0, 0);
            __builtin_prefetch(Xlo + pofs, 0, 0);
        }
        __syncthreads();

        // ---- 2 K-steps x 12 bf16 WMMAs: hi*hi + hi*lo + lo*hi ----
#pragma unroll
        for (int kk = 0; kk < KC2; kk += 32) {
            int ko = kk + h8;
            v16bf ah0 = ld_frag(XhiS, wm * 32 + l15, ko);
            v16bf ah1 = ld_frag(XhiS, wm * 32 + l15 + 16, ko);
            v16bf al0 = ld_frag(XloS, wm * 32 + l15, ko);
            v16bf al1 = ld_frag(XloS, wm * 32 + l15 + 16, ko);
            v16bf bh0 = ld_frag(WhiS, wn * 32 + l15, ko);
            v16bf bh1 = ld_frag(WhiS, wn * 32 + l15 + 16, ko);
            v16bf bl0 = ld_frag(WloS, wn * 32 + l15, ko);
            v16bf bl1 = ld_frag(WloS, wn * 32 + l15 + 16, ko);

            c00 = __builtin_amdgcn_wmma_f32_16x16x32_bf16(false, ah0, false, bh0, (short)0, c00, false, false);
            c01 = __builtin_amdgcn_wmma_f32_16x16x32_bf16(false, ah0, false, bh1, (short)0, c01, false, false);
            c10 = __builtin_amdgcn_wmma_f32_16x16x32_bf16(false, ah1, false, bh0, (short)0, c10, false, false);
            c11 = __builtin_amdgcn_wmma_f32_16x16x32_bf16(false, ah1, false, bh1, (short)0, c11, false, false);
            c00 = __builtin_amdgcn_wmma_f32_16x16x32_bf16(false, ah0, false, bl0, (short)0, c00, false, false);
            c01 = __builtin_amdgcn_wmma_f32_16x16x32_bf16(false, ah0, false, bl1, (short)0, c01, false, false);
            c10 = __builtin_amdgcn_wmma_f32_16x16x32_bf16(false, ah1, false, bl0, (short)0, c10, false, false);
            c11 = __builtin_amdgcn_wmma_f32_16x16x32_bf16(false, ah1, false, bl1, (short)0, c11, false, false);
            c00 = __builtin_amdgcn_wmma_f32_16x16x32_bf16(false, al0, false, bh0, (short)0, c00, false, false);
            c01 = __builtin_amdgcn_wmma_f32_16x16x32_bf16(false, al0, false, bh1, (short)0, c01, false, false);
            c10 = __builtin_amdgcn_wmma_f32_16x16x32_bf16(false, al1, false, bh0, (short)0, c10, false, false);
            c11 = __builtin_amdgcn_wmma_f32_16x16x32_bf16(false, al1, false, bh1, (short)0, c11, false, false);
        }
        __syncthreads();
    }

    // ---- epilogue: folded bias + store per C/D VGPR layout ----
    float w[BULK_DIM];
    fold_weights(w);
    const int ncol0 = n0 + wn * 32 + l15;
    const int ncol1 = ncol0 + 16;
    float bias0 = 0.0f, bias1 = 0.0f;
#pragma unroll
    for (int j = 0; j < BULK_DIM; ++j) {
        bias0 += w[j] * bvec[j * D_MODEL + ncol0];
        bias1 += w[j] * bvec[j * D_MODEL + ncol1];
    }
    const int mbase = m0 + wm * 32 + ((lane >> 4) << 3);
#pragma unroll
    for (int r = 0; r < 8; ++r) {
        int mr0 = mbase + r;
        Out[(size_t)mr0 * D_MODEL + ncol0] = c00[r] + bias0;
        Out[(size_t)mr0 * D_MODEL + ncol1] = c01[r] + bias1;
        int mr1 = mr0 + 16;
        Out[(size_t)mr1 * D_MODEL + ncol0] = c10[r] + bias0;
        Out[(size_t)mr1 * D_MODEL + ncol1] = c11[r] + bias1;
    }
}

// ===========================================================================
// Fallback (zero workspace): fp32 WMMA 16x16x4 GEMM with the 10-slice W-fold
// performed during LDS staging (W is L2-resident, re-reads are cheap).
// ===========================================================================
#define KC    32
#define LSTR  36
#define MT    128
#define NT    64

__global__ __launch_bounds__(256) void gemm_wmma_f32_fold_kernel(
    const float* __restrict__ X, const float* __restrict__ W,
    const float* __restrict__ bvec, float* __restrict__ Out) {

    __shared__ float Xs[MT * LSTR];
    __shared__ float Ws[NT * LSTR];

    float wf[BULK_DIM];
    fold_weights(wf);

    const int tid  = threadIdx.x;
    const int lane = tid & 31;
    const int wave = tid >> 5;
    const int wm   = wave & 3;
    const int wn   = wave >> 2;
    const int m0   = blockIdx.x * MT;
    const int n0   = blockIdx.y * NT;
    const int l15   = lane & 15;
    const int khalf = (lane >> 4) << 1;

    v8f c00 = {}, c01 = {}, c10 = {}, c11 = {};

    for (int k0 = 0; k0 < D_MODEL; k0 += KC) {
#pragma unroll
        for (int i = 0; i < 4; ++i) {
            int idx = tid + i * 256;
            int row = idx >> 3, c4 = idx & 7;
            v4f v = *(const v4f*)(X + (size_t)(m0 + row) * D_MODEL + k0 + c4 * 4);
            *(v4f*)(&Xs[row * LSTR + c4 * 4]) = v;
        }
        {
            int n  = tid & 63;
            int kq = tid >> 6;
            float tmp[8];
#pragma unroll
            for (int i = 0; i < 8; ++i) {
                size_t kb = (size_t)(k0 + kq * 8 + i) * (D_MODEL * BULK_DIM);
                float acc = 0.0f;
#pragma unroll
                for (int j = 0; j < BULK_DIM; ++j)
                    acc += wf[j] * W[kb + j * D_MODEL + n0 + n];
                tmp[i] = acc;
            }
            *(v4f*)(&Ws[n * LSTR + kq * 8])     = *(v4f*)(&tmp[0]);
            *(v4f*)(&Ws[n * LSTR + kq * 8 + 4]) = *(v4f*)(&tmp[4]);
        }
        __syncthreads();
#pragma unroll
        for (int kk = 0; kk < KC; kk += 4) {
            const float* xa = &Xs[(wm * 32 + l15) * LSTR + kk + khalf];
            v2f a0 = *(const v2f*)(xa);
            v2f a1 = *(const v2f*)(xa + 16 * LSTR);
            const float* wb = &Ws[(wn * 32 + l15) * LSTR + kk + khalf];
            v2f b0 = *(const v2f*)(wb);
            v2f b1 = *(const v2f*)(wb + 16 * LSTR);
            c00 = __builtin_amdgcn_wmma_f32_16x16x4_f32(false, a0, false, b0, (short)0, c00, false, false);
            c01 = __builtin_amdgcn_wmma_f32_16x16x4_f32(false, a0, false, b1, (short)0, c01, false, false);
            c10 = __builtin_amdgcn_wmma_f32_16x16x4_f32(false, a1, false, b0, (short)0, c10, false, false);
            c11 = __builtin_amdgcn_wmma_f32_16x16x4_f32(false, a1, false, b1, (short)0, c11, false, false);
        }
        __syncthreads();
    }

    float w[BULK_DIM];
    fold_weights(w);
    const int ncol0 = n0 + wn * 32 + l15;
    const int ncol1 = ncol0 + 16;
    float bias0 = 0.0f, bias1 = 0.0f;
#pragma unroll
    for (int j = 0; j < BULK_DIM; ++j) {
        bias0 += w[j] * bvec[j * D_MODEL + ncol0];
        bias1 += w[j] * bvec[j * D_MODEL + ncol1];
    }
    const int mbase = m0 + wm * 32 + ((lane >> 4) << 3);
#pragma unroll
    for (int r = 0; r < 8; ++r) {
        int mr0 = mbase + r;
        Out[(size_t)mr0 * D_MODEL + ncol0] = c00[r] + bias0;
        Out[(size_t)mr0 * D_MODEL + ncol1] = c01[r] + bias1;
        int mr1 = mr0 + 16;
        Out[(size_t)mr1 * D_MODEL + ncol0] = c10[r] + bias0;
        Out[(size_t)mr1 * D_MODEL + ncol1] = c11[r] + bias1;
    }
}

// ===========================================================================
extern "C" void kernel_launch(void* const* d_in, const int* in_sizes, int n_in,
                              void* d_out, int out_size, void* d_ws, size_t ws_size,
                              hipStream_t stream) {
    const float* X    = (const float*)d_in[0];   // (4,2048,1024) fp32
    const float* W    = (const float*)d_in[1];   // (1024,10240)  fp32
    const float* bvec = (const float*)d_in[2];   // (10240,)      fp32
    float*       Out  = (float*)d_out;

    const size_t W_ELEMS = (size_t)D_MODEL * D_MODEL;   // 1M
    const size_t X_ELEMS = (size_t)M_TOTAL * D_MODEL;   // 8M
    const size_t need    = (2 * W_ELEMS + 2 * X_ELEMS) * sizeof(unsigned short); // 36 MB

    dim3 grid(M_TOTAL / 128, D_MODEL / 64);  // 64 x 16

    if (ws_size >= need) {
        unsigned short* base = (unsigned short*)d_ws;
        unsigned short* WhiT = base;
        unsigned short* WloT = base + W_ELEMS;
        unsigned short* XhiP = base + 2 * W_ELEMS;
        unsigned short* XloP = base + 2 * W_ELEMS + X_ELEMS;

        dim3 gW(D_MODEL / 64, D_MODEL / 64); // 16 x 16
        fold_split_w_kernel<<<gW, 256, 0, stream>>>(W, WhiT, WloT);
        split_x_kernel<<<(int)(X_ELEMS / 8 / 256), 256, 0, stream>>>(X, XhiP, XloP);
        gemm_bf16x3_kernel<<<grid, 256, 0, stream>>>(XhiP, XloP, WhiT, WloT, bvec, Out);
    } else {
        gemm_wmma_f32_fold_kernel<<<grid, 256, 0, stream>>>(X, W, bvec, Out);
    }
}
